// AxialAttention_72275709657477
// MI455X (gfx1250) — matrix operand order repaired and checked by reading, not compiled, not run on hardware
//
#include <hip/hip_runtime.h>

typedef __attribute__((ext_vector_type(16))) __bf16 v16bf;
typedef __attribute__((ext_vector_type(8)))  __bf16 v8bf;
typedef __attribute__((ext_vector_type(8)))  float  v8f;

// ---------------- constants ----------------
#define NB 16
#define NC 256
#define NH 64
#define NW 64
#define NWN 1024           // NB*NW batch-fold
#define THR 768            // 3*OUT
#define OUTC 256
#define NG 8
#define GP 32

// workspace byte offsets (all 256B aligned)
#define OFF_QSUM    0u            // 768 f32
#define OFF_QSQ     3072u
#define OFF_QSCALE  6144u
#define OFF_QSHIFT  9216u
#define OFF_SSUM    12288u        // 8 f32
#define OFF_SSQ     12544u
#define OFF_SSCALE  12800u
#define OFF_SSHIFT  13056u
#define OFF_OSUM    13312u        // 256 f32
#define OFF_OSQ     14336u
#define OFF_OSCALE  15360u
#define OFF_OSHIFT  16384u
#define OFF_WBF     32768u                         // 768*256 bf16
#define OFF_QKV     524288u                        // 1024*768*64 bf16 = 100,663,296
#define OFF_QK      101187584u                     // 1024*8*64*64 f32 = 134,217,728
#define OFF_SV      235405312u                     // 1024*256*64 f32  = 67,108,864

__device__ __forceinline__ v8f zero8() {
  v8f z = {0.f, 0.f, 0.f, 0.f, 0.f, 0.f, 0.f, 0.f};
  return z;
}

// A fragment: lane holds row m, K-run [half*16,+16)
__device__ __forceinline__ v16bf load_a_frag(const __bf16* row, int hf) {
  v8bf lo = *reinterpret_cast<const v8bf*>(row + hf * 16);
  v8bf hi = *reinterpret_cast<const v8bf*>(row + hf * 16 + 8);
  return __builtin_shufflevector(lo, hi, 0, 1, 2, 3, 4, 5, 6, 7, 8, 9, 10, 11, 12, 13, 14, 15);
}

__device__ __forceinline__ float half_reduce(float v) { // sum over 16 lanes of a half-wave
  v += __shfl_xor(v, 1, 32);
  v += __shfl_xor(v, 2, 32);
  v += __shfl_xor(v, 4, 32);
  v += __shfl_xor(v, 8, 32);
  return v;
}
__device__ __forceinline__ float wave_reduce(float v) {
  v = half_reduce(v);
  v += __shfl_xor(v, 16, 32);
  return v;
}

// ---------------- kernel 0: zero stats + convert w to bf16 ----------------
__global__ __launch_bounds__(256) void k_init(const float* __restrict__ w,
                                              __bf16* __restrict__ wbf,
                                              float* __restrict__ stats) {
  const int idx = blockIdx.x * 256 + threadIdx.x;
  if (idx < 4352) stats[idx] = 0.f;                 // covers all stat arrays
  if (idx < THR * NC) wbf[idx] = (__bf16)w[idx];
}

// ---------------- kernel 1: QKV GEMM + per-channel stats ----------------
// out[o][h] = sum_c w[o][c] * x[b,c,h,w]   for n=(b,w)
__global__ __launch_bounds__(256) void k_qkv(const float* __restrict__ x,
                                             const __bf16* __restrict__ wbf,
                                             __bf16* __restrict__ qkv,
                                             float* __restrict__ gsum,
                                             float* __restrict__ gsq) {
  __shared__ alignas(64) __bf16 Bs[64][272];        // [h][c] bf16, pitch 544B (32B mult)
  __shared__ float s_sum[THR], s_sq[THR];
  const int t = threadIdx.x;
  const int n = blockIdx.x, b = n >> 6, w = n & 63;
  for (int i = t; i < THR; i += 256) { s_sum[i] = 0.f; s_sq[i] = 0.f; }
  const float* xb = x + (size_t)b * NC * NH * NW + w;
  for (int idx = t; idx < NC * NH; idx += 256) {
    const int c = idx >> 6, h = idx & 63;
    Bs[h][c] = (__bf16)xb[(size_t)(c * NH + h) * NW];
  }
  __syncthreads();
  const int wave = t >> 5, lane = t & 31, hf = lane >> 4, l16 = lane & 15;
  for (int mt = wave; mt < 48; mt += 8) {           // uniform per wave
    const int m0 = mt << 4;
    v8f acc[4];
#pragma unroll
    for (int nt = 0; nt < 4; ++nt) acc[nt] = zero8();
#pragma unroll
    for (int kk = 0; kk < 8; ++kk) {
      const __bf16* ap = wbf + (m0 + l16) * NC + kk * 32;
      v16bf a = load_a_frag(ap, hf);
#pragma unroll
      for (int nt = 0; nt < 4; ++nt) {
        v16bf bb = *reinterpret_cast<const v16bf*>(&Bs[nt * 16 + l16][kk * 32 + hf * 16]);
        acc[nt] = __builtin_amdgcn_wmma_f32_16x16x32_bf16(false, a, false, bb, (short)0,
                                                          acc[nt], false, false);
      }
    }
#pragma unroll
    for (int nt = 0; nt < 4; ++nt) {
      const int h = nt * 16 + l16;
#pragma unroll
      for (int r = 0; r < 8; ++r) {
        const int o = m0 + r + hf * 8;
        const float v = acc[nt][r];
        qkv[((size_t)n * THR + o) * NH + h] = (__bf16)v;
        const float s = half_reduce(v);
        const float q = half_reduce(v * v);
        if (l16 == 0) { atomicAdd(&s_sum[o], s); atomicAdd(&s_sq[o], q); }
      }
    }
  }
  __syncthreads();
  for (int i = t; i < THR; i += 256) {
    atomicAdd(&gsum[i], s_sum[i]);
    atomicAdd(&gsq[i], s_sq[i]);
  }
}

// ---------------- generic BN finalize: scale/shift from sum/sumsq ----------------
__global__ __launch_bounds__(256) void k_finalize(const float* __restrict__ gsum,
                                                  const float* __restrict__ gsq,
                                                  const float* __restrict__ gamma,
                                                  const float* __restrict__ beta,
                                                  float* __restrict__ scale,
                                                  float* __restrict__ shift,
                                                  int count, float invN) {
  const int o = blockIdx.x * 256 + threadIdx.x;
  if (o >= count) return;
  const float m = gsum[o] * invN;
  const float var = gsq[o] * invN - m * m;
  const float sc = gamma[o] * rsqrtf(var + 1e-5f);
  scale[o] = sc;
  shift[o] = beta[o] - m * sc;
}

// ---------------- kernel 3: qk = q^T k per (n,g) + group stats ----------------
__global__ __launch_bounds__(256) void k_qk(const __bf16* __restrict__ qkv,
                                            const float* __restrict__ qscale,
                                            const float* __restrict__ qshift,
                                            float* __restrict__ qk,
                                            float* __restrict__ gssum,
                                            float* __restrict__ gssq) {
  __shared__ alignas(64) __bf16 kt[64][256];        // [j][c] affined K, pitch 512B
  __shared__ float s_qs[256], s_qsh[256];
  const int t = threadIdx.x, n = blockIdx.x;
  s_qs[t] = qscale[t];
  s_qsh[t] = qshift[t];
  for (int idx = t; idx < 256 * NH; idx += 256) {
    const int o = idx >> 6, h = idx & 63;
    const float v = (float)qkv[((size_t)n * THR + 256 + o) * NH + h];
    kt[h][o] = (__bf16)(v * qscale[256 + o] + qshift[256 + o]);
  }
  __syncthreads();
  const int g = t >> 5, lane = t & 31, hf = lane >> 4, l16 = lane & 15;
  const __bf16* qbase = qkv + (size_t)n * THR * NH;
  float lsum = 0.f, lsq = 0.f;
#pragma unroll
  for (int mt = 0; mt < 4; ++mt) {
    const int i = mt * 16 + l16;
    v16bf a;                                        // A[m=i][k=c] = affine(q[c][i])
#pragma unroll
    for (int e = 0; e < 16; ++e) {
      const int c = hf * 16 + e;
      const int o = g * GP + c;
      const float v = (float)qbase[(size_t)o * NH + i];
      a[e] = (__bf16)(v * s_qs[o] + s_qsh[o]);
    }
#pragma unroll
    for (int nt = 0; nt < 4; ++nt) {
      v16bf bb = *reinterpret_cast<const v16bf*>(&kt[nt * 16 + l16][g * GP + hf * 16]);
      v8f acc = __builtin_amdgcn_wmma_f32_16x16x32_bf16(false, a, false, bb, (short)0,
                                                        zero8(), false, false);
      const int j = nt * 16 + l16;
#pragma unroll
      for (int r = 0; r < 8; ++r) {
        const int ii = mt * 16 + r + hf * 8;
        const float v = acc[r];
        qk[(((size_t)n * NG + g) * NH + ii) * NH + j] = v;
        lsum += v;
        lsq += v * v;
      }
    }
  }
  lsum = wave_reduce(lsum);
  lsq = wave_reduce(lsq);
  if (lane == 0) { atomicAdd(&gssum[g], lsum); atomicAdd(&gssq[g], lsq); }
}

// ---------------- kernel 5: softmax + sv = sim @ v + out-channel stats ----------------
// block handles (n, 4 groups): grid = 2048
__global__ __launch_bounds__(256) void k_attn(const __bf16* __restrict__ qkv,
                                              const float* __restrict__ qscale,
                                              const float* __restrict__ qshift,
                                              const float* __restrict__ qk,
                                              const float* __restrict__ sscale,
                                              const float* __restrict__ sshift,
                                              float* __restrict__ sv,
                                              float* __restrict__ gosum,
                                              float* __restrict__ gosq) {
  __shared__ alignas(64) __bf16 s_sim[4][64][80];   // [g][i][j] pitch 160B (32B mult)
  __shared__ float s_osum[128], s_osq[128];
  const int t = threadIdx.x;
  const int n = blockIdx.x >> 1, gbase = (blockIdx.x & 1) * 4;
  if (t < 128) { s_osum[t] = 0.f; s_osq[t] = 0.f; }
  {                                                 // one softmax row per thread
    const int glr = t >> 6, i = t & 63, g = gbase + glr;
    const float sc = sscale[g], sh = sshift[g];
    const float* row = qk + (((size_t)n * NG + g) * NH + i) * NH;
    float ev[64];
    float mx = -3.4e38f;
#pragma unroll
    for (int j = 0; j < 64; ++j) { const float v = row[j] * sc + sh; ev[j] = v; mx = fmaxf(mx, v); }
    float sum = 0.f;
#pragma unroll
    for (int j = 0; j < 64; ++j) { const float e = __expf(ev[j] - mx); ev[j] = e; sum += e; }
    const float inv = 1.f / sum;
#pragma unroll
    for (int j = 0; j < 64; ++j) s_sim[glr][i][j] = (__bf16)(ev[j] * inv);
  }
  __syncthreads();
  const int wv = t >> 5, lane = t & 31, hf = lane >> 4, l16 = lane & 15;
  const int gl = wv >> 1, g = gbase + gl, mt = wv & 1;
  const int orow = 2 * OUTC + g * GP + mt * 16 + l16;   // v channel for this A row
  const __bf16* vrow = qkv + ((size_t)n * THR + orow) * NH;
  const float vs = qscale[orow], vh = qshift[orow];
  v8f acc[4];
#pragma unroll
  for (int nt = 0; nt < 4; ++nt) acc[nt] = zero8();
#pragma unroll
  for (int ks = 0; ks < 64; ks += 32) {
    v16bf a = load_a_frag(vrow + ks, hf);           // A[m=c][k=j] = affine(v[c][j])
#pragma unroll
    for (int e = 0; e < 16; ++e) a[e] = (__bf16)((float)a[e] * vs + vh);
#pragma unroll
    for (int nt = 0; nt < 4; ++nt) {
      v16bf bb = *reinterpret_cast<const v16bf*>(&s_sim[gl][nt * 16 + l16][ks + hf * 16]);
      acc[nt] = __builtin_amdgcn_wmma_f32_16x16x32_bf16(false, a, false, bb, (short)0,
                                                        acc[nt], false, false);
    }
  }
#pragma unroll
  for (int nt = 0; nt < 4; ++nt) {
    const int i = nt * 16 + l16;
#pragma unroll
    for (int r = 0; r < 8; ++r) {
      const int cc = g * GP + mt * 16 + r + hf * 8; // 0..255 global out channel
      const float v = acc[nt][r];
      sv[((size_t)n * OUTC + cc) * NH + i] = v;
      const float s = half_reduce(v);
      const float q = half_reduce(v * v);
      if (l16 == 0) {
        const int ol = cc - gbase * GP;             // 0..127 within block
        atomicAdd(&s_osum[ol], s);
        atomicAdd(&s_osq[ol], q);
      }
    }
  }
  __syncthreads();
  if (t < 128) {
    atomicAdd(&gosum[gbase * GP + t], s_osum[t]);
    atomicAdd(&gosq[gbase * GP + t], s_osq[t]);
  }
}

// ---------------- kernel 7: out BN affine + transpose [n,o,h] -> [b,o,h,w] ----------------
__global__ __launch_bounds__(256) void k_out(const float* __restrict__ sv,
                                             const float* __restrict__ oscale,
                                             const float* __restrict__ oshift,
                                             float* __restrict__ out) {
  const size_t idx = (size_t)blockIdx.x * 256 + threadIdx.x;
  const int w = (int)(idx & 63);
  const int h = (int)((idx >> 6) & 63);
  const int o = (int)((idx >> 12) & 255);
  const int b = (int)(idx >> 20);
  const float v = sv[(((size_t)b * NW + w) * OUTC + o) * NH + h];
  out[idx] = v * oscale[o] + oshift[o];
}

// ---------------- launcher ----------------
extern "C" void kernel_launch(void* const* d_in, const int* in_sizes, int n_in,
                              void* d_out, int out_size, void* d_ws, size_t ws_size,
                              hipStream_t stream) {
  (void)in_sizes; (void)n_in; (void)out_size; (void)ws_size;
  const float* x         = (const float*)d_in[0];
  const float* w_qkv     = (const float*)d_in[1];
  const float* gamma_qkv = (const float*)d_in[2];
  const float* beta_qkv  = (const float*)d_in[3];
  const float* gamma_sim = (const float*)d_in[4];
  const float* beta_sim  = (const float*)d_in[5];
  const float* gamma_out = (const float*)d_in[6];
  const float* beta_out  = (const float*)d_in[7];
  float* out = (float*)d_out;

  char* ws = (char*)d_ws;
  float* g_qsum   = (float*)(ws + OFF_QSUM);
  float* g_qsq    = (float*)(ws + OFF_QSQ);
  float* g_qscale = (float*)(ws + OFF_QSCALE);
  float* g_qshift = (float*)(ws + OFF_QSHIFT);
  float* g_ssum   = (float*)(ws + OFF_SSUM);
  float* g_ssq    = (float*)(ws + OFF_SSQ);
  float* g_sscale = (float*)(ws + OFF_SSCALE);
  float* g_sshift = (float*)(ws + OFF_SSHIFT);
  float* g_osum   = (float*)(ws + OFF_OSUM);
  float* g_osq    = (float*)(ws + OFF_OSQ);
  float* g_oscale = (float*)(ws + OFF_OSCALE);
  float* g_oshift = (float*)(ws + OFF_OSHIFT);
  __bf16* wbf     = (__bf16*)(ws + OFF_WBF);
  __bf16* qkv     = (__bf16*)(ws + OFF_QKV);
  float* qk       = (float*)(ws + OFF_QK);
  float* sv       = (float*)(ws + OFF_SV);

  k_init<<<dim3(768), dim3(256), 0, stream>>>(w_qkv, wbf, (float*)ws);
  k_qkv<<<dim3(NWN), dim3(256), 0, stream>>>(x, wbf, qkv, g_qsum, g_qsq);
  k_finalize<<<dim3(3), dim3(256), 0, stream>>>(g_qsum, g_qsq, gamma_qkv, beta_qkv,
                                                g_qscale, g_qshift, THR, 1.f / 65536.f);
  k_qk<<<dim3(NWN), dim3(256), 0, stream>>>(qkv, g_qscale, g_qshift, qk, g_ssum, g_ssq);
  k_finalize<<<dim3(1), dim3(256), 0, stream>>>(g_ssum, g_ssq, gamma_sim, beta_sim,
                                                g_sscale, g_sshift, NG, 1.f / 4194304.f);
  k_attn<<<dim3(NWN * 2), dim3(256), 0, stream>>>(qkv, g_qscale, g_qshift, qk,
                                                  g_sscale, g_sshift, sv, g_osum, g_osq);
  k_finalize<<<dim3(1), dim3(256), 0, stream>>>(g_osum, g_osq, gamma_out, beta_out,
                                                g_oscale, g_oshift, OUTC, 1.f / 65536.f);
  k_out<<<dim3(65536), dim3(256), 0, stream>>>(sv, g_oscale, g_oshift, out);
}
